// SimilarityTripletLoss_54606214201624
// MI455X (gfx1250) — matrix operand order repaired and checked by reading, not compiled, not run on hardware
//
#include <hip/hip_runtime.h>
#include <hip/hip_bf16.h>

#define HWP   4096           // H*W = P = Q
#define CCH   256            // channels
#define NB    4              // batch
#define WDIM  64
#define HDIM  64
#define MARG  4.1f
#define TOPK  100

typedef __attribute__((ext_vector_type(16))) _Float16 v16h;
typedef __attribute__((ext_vector_type(8)))  _Float16 v8h;
typedef __attribute__((ext_vector_type(8)))  float    v8f;
typedef __attribute__((ext_vector_type(4)))  float    fv4;

// ---------------------------------------------------------------------------
// Kernel 1: positive-pair selection + normalization + f16 packing + exact
// f32 d_pos, Graw, pair_valid.  One thread per pair index p.
// ---------------------------------------------------------------------------
__global__ __launch_bounds__(256)
void prep_kernel(const float* __restrict__ sketch,  // [B,C,H,W]
                 const float* __restrict__ ref,     // [B,C,H,W]
                 const float* __restrict__ G,       // [B,H,W,2]
                 _Float16* __restrict__ Abuf,       // [B,P,C] anchor (unit, f16)
                 _Float16* __restrict__ Bbuf,       // [B,Q,C] ref    (unit, f16)
                 float*    __restrict__ dpos,       // [B,P]
                 int*      __restrict__ pval,       // [B,P]
                 float*    __restrict__ graw)       // [B,P,2]
{
  int tid = blockIdx.x * blockDim.x + threadIdx.x;
  if (tid >= NB * HWP) return;
  int b = tid / HWP;
  int p = tid - b * HWP;

  float Gx = G[(size_t)tid * 2 + 0];
  float Gy = G[(size_t)tid * 2 + 1];
  float gx = Gx * (float)WDIM;
  float gy = Gy * (float)HDIM;

  float x0 = floorf(gx), y0 = floorf(gy);
  float cx[4] = {x0, x0 + 1.f, x0 + 1.f, x0};
  float cy[4] = {y0, y0, y0 + 1.f, y0 + 1.f};
  float dst[4]; bool val[4]; float maxd = -1.f;
  #pragma unroll
  for (int j = 0; j < 4; ++j) {
    float dx = gx - cx[j], dy = gy - cy[j];
    dst[j] = sqrtf(dx * dx + dy * dy);
    val[j] = (cx[j] >= 0.f) && (cy[j] >= 0.f) &&
             (cx[j] <= (float)(WDIM - 1)) && (cy[j] <= (float)(HDIM - 1));
    maxd = fmaxf(maxd, dst[j]);
  }
  int   sel = 0; float best = -3.0e38f; bool anyv = false;
  #pragma unroll
  for (int j = 0; j < 4; ++j) {
    float pr = val[j] ? (maxd + 0.3f - dst[j]) : -3.0e38f;
    if (pr > best) { best = pr; sel = j; }   // first max (argmax semantics)
    anyv = anyv || val[j];
  }
  int ax = (int)cx[sel], ay = (int)cy[sel];
  int aflat = ay * WDIM + ax;
  aflat = aflat < 0 ? 0 : (aflat > HWP - 1 ? HWP - 1 : aflat);

  // pass 1: squared norms of sketch@aflat and ref@p (stride-HWP reads, coalesced)
  const float* skb = sketch + (size_t)b * CCH * HWP + aflat;
  const float* rfb = ref    + (size_t)b * CCH * HWP + p;
  float ssA = 0.f, ssR = 0.f;
  for (int c = 0; c < CCH; ++c) {
    float va = skb[(size_t)c * HWP];
    float vr = rfb[(size_t)c * HWP];
    ssA += va * va; ssR += vr * vr;
  }
  float invA = 1.f / fmaxf(sqrtf(ssA), 1e-8f);
  float invR = 1.f / fmaxf(sqrtf(ssR), 1e-8f);

  // pass 2: write f16 unit vectors, accumulate exact f32 dot for d_pos
  _Float16* Ao = Abuf + (size_t)tid * CCH;
  _Float16* Bo = Bbuf + (size_t)tid * CCH;
  float dot = 0.f;
  for (int c = 0; c < CCH; ++c) {
    float va = skb[(size_t)c * HWP] * invA;
    float vr = rfb[(size_t)c * HWP] * invR;
    Ao[c] = (_Float16)va;
    Bo[c] = (_Float16)vr;
    dot += va * vr;
  }
  dpos[tid] = 1.f - dot;
  pval[tid] = anyv ? 1 : 0;
  graw[(size_t)tid * 2 + 0] = Gx * (float)(WDIM * WDIM);   // Gxy * scale (scaled twice)
  graw[(size_t)tid * 2 + 1] = Gy * (float)(HDIM * HDIM);
}

// wave32 reductions -----------------------------------------------------------
__device__ __forceinline__ int wred_add_i(int v) {
  #pragma unroll
  for (int s = 16; s > 0; s >>= 1) v += __shfl_xor(v, s, 32);
  return v;
}
__device__ __forceinline__ float wred_add_f(float v) {
  #pragma unroll
  for (int s = 16; s > 0; s >>= 1) v += __shfl_xor(v, s, 32);
  return v;
}

// ---------------------------------------------------------------------------
// Kernel 2: fused WMMA GEMM (16-row block vs all 4096 refs) into 256 KB LDS,
// then per-row top-100 mining + filter + accumulate.
// Block = 512 threads = 16 waves.  Grid = (P/16, B).
// ---------------------------------------------------------------------------
__global__ __launch_bounds__(512)
void gemm_mine_kernel(const _Float16* __restrict__ Abuf,
                      const _Float16* __restrict__ Bbuf,
                      const float*    __restrict__ dpos,
                      const int*      __restrict__ pval,
                      const float*    __restrict__ graw,
                      float*          __restrict__ acc)   // [2]: sum, n
{
  extern __shared__ float lds[];                // 16 * 4096 f32 = 256 KB
  const int b     = blockIdx.y;
  const int pBase = blockIdx.x * 16;
  const int wave  = threadIdx.x >> 5;
  const int lane  = threadIdx.x & 31;
  const int hi    = lane >> 4;                  // lane group
  const int l15   = lane & 15;

  // ---- hoist the 8 A fragments (this wave's 16x256 row block) -------------
  // A k order used here (contiguous, same as B): lane<16 holds K 0..15; lane>=16 K 16..31
  const _Float16* Arow = Abuf + ((size_t)(b * HWP + pBase + l15)) * CCH;
  v16h afrag[8];
  #pragma unroll
  for (int kk = 0; kk < 8; ++kk) {
    const int kB = kk * 32;
    afrag[kk] = *(const v16h*)(Arow + kB + hi * 16);
  }

  // ---- GEMM phase: wave w owns column tiles w, w+16, ... ------------------
  const _Float16* Bbase = Bbuf + (size_t)b * HWP * CCH + hi * 16;
  for (int t = wave; t < HWP / 16; t += 16) {
    const int qBase = t * 16;
    // prefetch next tile's B row for this lane (global_prefetch_b8)
    {
      int tn = t + 16; tn = tn < HWP / 16 ? tn : t;
      const char* pf = (const char*)(Bbase + ((size_t)(tn * 16 + l15)) * CCH);
      __builtin_prefetch(pf, 0, 3);
      __builtin_prefetch(pf + 256, 0, 3);
    }
    // B k order used here (contiguous, same as A): lane<16 col=lane K 0..15; lane>=16 col=lane-16 K 16..31
    const _Float16* Brow = Bbase + ((size_t)(qBase + l15)) * CCH;
    v8f c = {};
    #pragma unroll
    for (int kk = 0; kk < 8; ++kk) {
      v16h bfrag = *(const v16h*)(Brow + kk * 32);
      c = __builtin_amdgcn_wmma_f32_16x16x32_f16(false, afrag[kk], false, bfrag,
                                                 (short)0, c, false, false);
    }
    // C/D layout: vgpr r, lane<16 -> row r; lane>=16 -> row r+8; col = lane%16
    #pragma unroll
    for (int r = 0; r < 8; ++r)
      lds[(r + hi * 8) * HWP + qBase + l15] = c[r];
  }
  __syncthreads();

  // ---- mining phase: wave w mines row w -----------------------------------
  const int   gp   = pBase + wave;
  const float dp   = dpos[b * HWP + gp];
  const float addc = dp + (MARG - 1.0f);        // loss = cos + addc  (> 0)
  const float* row = lds + wave * HWP;
  const float gpx = graw[((size_t)b * HWP + gp) * 2 + 0];
  const float gpy = graw[((size_t)b * HWP + gp) * 2 + 1];

  // pass 0: row min/max (vectorized ds_load_b128) to seed the bit search
  float vmin = 3.0e38f, vmax = 0.f;
  for (int i = 0; i < 32; ++i) {
    fv4 v4 = *(const fv4*)(row + (i * 32 + lane) * 4);
    #pragma unroll
    for (int e = 0; e < 4; ++e) {
      float v = fmaxf(v4[e] + addc, 0.f);
      vmin = fminf(vmin, v); vmax = fmaxf(vmax, v);
    }
  }
  #pragma unroll
  for (int s = 16; s > 0; s >>= 1) {
    vmin = fminf(vmin, __shfl_xor(vmin, s, 32));
    vmax = fmaxf(vmax, __shfl_xor(vmax, s, 32));
  }

  // binary search on IEEE bits (positive floats are order-monotonic):
  // invariant cnt(lo) < TOPK <= cnt(hiB)
  unsigned lo = __float_as_uint(vmin);
  unsigned hiB = __float_as_uint(vmax) + 1u;
  int guard = 0;
  while (hiB - lo > 1u && guard++ < 40) {
    unsigned mid = lo + ((hiB - lo) >> 1);
    int cl = 0;
    for (int i = 0; i < 32; ++i) {
      fv4 v4 = *(const fv4*)(row + (i * 32 + lane) * 4);
      #pragma unroll
      for (int e = 0; e < 4; ++e) {
        float v = fmaxf(v4[e] + addc, 0.f);
        cl += (__float_as_uint(v) < mid) ? 1 : 0;
      }
    }
    cl = wred_add_i(cl);
    if (cl >= TOPK) hiB = mid; else lo = mid;
  }
  const unsigned kth = hiB - 1u;                // bits of the 100th smallest

  // pass A: strict-less elements -> rank count + filtered accumulation
  int cl = 0; float sum = 0.f, kc = 0.f;
  for (int i = 0; i < 32; ++i) {
    int qb = (i * 32 + lane) * 4;
    fv4 v4 = *(const fv4*)(row + qb);
    #pragma unroll
    for (int e = 0; e < 4; ++e) {
      float v = fmaxf(v4[e] + addc, 0.f);
      if (__float_as_uint(v) < kth) {
        ++cl;
        int q = qb + e;
        bool keep = (q != gp);
        float nx = graw[((size_t)b * HWP + q) * 2 + 0];
        float ny = graw[((size_t)b * HWP + q) * 2 + 1];
        if (fabsf(nx - gpx) < 1.2f && fabsf(ny - gpy) < 1.2f) keep = false;
        if (keep) { sum += v; kc += 1.f; }
      }
    }
  }
  int need = TOPK - wred_add_i(cl);              // >= 1 ties to take, in q order

  // pass B: ties at kth, taken in ascending q (argsort-stable)
  int taken = 0;
  for (int i = 0; i < 128 && taken < need; ++i) {
    int q = i * 32 + lane;
    float v = fmaxf(row[q] + addc, 0.f);
    bool tie = (__float_as_uint(v) == kth);
    unsigned mask = (unsigned)__ballot(tie);
    if (tie) {
      int rank = __popc(mask & ((1u << lane) - 1u));
      if (taken + rank < need) {
        bool keep = (q != gp);
        float nx = graw[((size_t)b * HWP + q) * 2 + 0];
        float ny = graw[((size_t)b * HWP + q) * 2 + 1];
        if (fabsf(nx - gpx) < 1.2f && fabsf(ny - gpy) < 1.2f) keep = false;
        if (keep) { sum += v; kc += 1.f; }
      }
    }
    taken += __popc(mask);
  }

  sum = wred_add_f(sum);
  kc  = wred_add_f(kc);

  if (lane == 0) {
    bool valid = (pval[b * HWP + gp] != 0) && (kc > 0.f);
    if (valid) {
      atomicAdd(acc + 0, sum / kc);
      atomicAdd(acc + 1, 1.0f);
    }
  }
}

// ---------------------------------------------------------------------------
__global__ void finalize_kernel(const float* __restrict__ acc, float* __restrict__ out)
{
  out[0] = acc[0] / (1e-6f + acc[1]);
}

// ---------------------------------------------------------------------------
extern "C" void kernel_launch(void* const* d_in, const int* in_sizes, int n_in,
                              void* d_out, int out_size, void* d_ws, size_t ws_size,
                              hipStream_t stream) {
  const float* sketch = (const float*)d_in[0];
  const float* refv   = (const float*)d_in[1];
  const float* G      = (const float*)d_in[2];
  float* out = (float*)d_out;

  char* ws = (char*)d_ws;
  float*    acc  = (float*)ws;                                   //   8 B (+pad)
  float*    dpos = (float*)(ws + 256);                           //  64 KB
  int*      pv   = (int*)  (ws + 256 + 65536);                   //  64 KB
  float*    grw  = (float*)(ws + 256 + 65536 + 65536);           // 128 KB
  _Float16* Ab   = (_Float16*)(ws + 262400);                     //   8 MB
  _Float16* Bb   = Ab + (size_t)NB * HWP * CCH;                  //   8 MB

  (void)in_sizes; (void)n_in; (void)out_size; (void)ws_size;

  hipMemsetAsync(acc, 0, 2 * sizeof(float), stream);

  prep_kernel<<<(NB * HWP + 255) / 256, 256, 0, stream>>>(
      sketch, refv, G, Ab, Bb, dpos, pv, grw);

  dim3 grid(HWP / 16, NB);
  size_t ldsBytes = (size_t)16 * HWP * sizeof(float);            // 256 KB / WGP
  gemm_mine_kernel<<<grid, 512, ldsBytes, stream>>>(Ab, Bb, dpos, pv, grw, acc);

  finalize_kernel<<<1, 1, 0, stream>>>(acc, out);
}
